// _NonLocalBlockND_56882546868763
// MI455X (gfx1250) — matrix operand order repaired and checked by reading, not compiled, not run on hardware
//
#include <hip/hip_runtime.h>
#include <stdint.h>

// ---------------- problem constants (from reference) ----------------
#define B_   4
#define C_   256
#define HW_  9216      // 96*96 spatial
#define CI_  128
#define HH_  96
#define PW_  47        // pooled width: floor((96-3)/2)+1
#define M_   2209      // 47*47
#define MP_  2240      // M padded to multiple of 32
#define NT16 (HW_/16)  // 576 row-blocks of 16

#define ATTN_LDS (16*MP_*4 + 16*CI_*2 + 16*32*2 + 32*4)  // 148608 B

typedef __bf16 bf16_t;
typedef __attribute__((ext_vector_type(16))) __bf16 v16bf;
typedef __attribute__((ext_vector_type(8)))  __bf16 v8bf;
typedef __attribute__((ext_vector_type(8)))  float  v8f;

// ---------------- optional CDNA5 async global->LDS path ----------------
#if __has_builtin(__builtin_amdgcn_global_load_async_to_lds_b128)
#define USE_ASYNC_LDS 1
typedef int v4i_ __attribute__((vector_size(16)));
typedef __attribute__((address_space(1))) v4i_ as1_v4i;
typedef __attribute__((address_space(3))) v4i_ as3_v4i;
#define TO_AS1(p) ((as1_v4i*)(uintptr_t)(p))
#define TO_AS3(p) ((as3_v4i*)(unsigned)(uintptr_t)(p))
__device__ inline void wait_async0() {
#if __has_builtin(__builtin_amdgcn_s_wait_asynccnt)
  __builtin_amdgcn_s_wait_asynccnt(0);
#else
  asm volatile("s_wait_asynccnt 0" ::: "memory");
#endif
}
#else
#define USE_ASYNC_LDS 0
#endif

// ---------------- WMMA helpers (CDNA5 16x16x32 bf16, f32 accum) ----------------
__device__ inline v8f WMMA(v16bf a, v16bf b, v8f c) {
  // D = A(16x32) * B(32x16) + C ; emits v_wmma_f32_16x16x32_bf16
  return __builtin_amdgcn_wmma_f32_16x16x32_bf16(
      /*neg_a=*/false, a, /*neg_b=*/false, b,
      /*c_mod=*/(short)0, c, /*reuse_a=*/false, /*reuse_b=*/false);
}

// A fragment: 16x32 bf16 tile at row-major src (leading dim ld elements).
// Lane L holds row L%16; lanes 0-15: K 0..7 (v0-3) & 16..23 (v4-7); lanes 16-31: K 8..15 & 24..31.
__device__ inline v16bf wmma_load_a(const bf16_t* src, int ld) {
  unsigned lane = threadIdx.x & 31u;
  unsigned r = lane & 15u;
  unsigned k = (lane < 16u) ? 0u : 8u;
  const bf16_t* p = src + (size_t)r * ld + k;
  v8bf lo = *(const v8bf*)(p);
  v8bf hi = *(const v8bf*)(p + 16);
  return __builtin_shufflevector(lo, hi, 0,1,2,3,4,5,6,7,8,9,10,11,12,13,14,15);
}

// B fragment: 32x16 bf16 (KxN). src is row-major [N rows x K] ("Bt", K contiguous).
// Lane L holds column L%16; lanes 0-15: K 0..7 & 16..23; lanes 16-31: K 8..15 & 24..31.
__device__ inline v16bf wmma_load_b(const bf16_t* src, int ld) {
  unsigned lane = threadIdx.x & 31u;
  unsigned n = lane & 15u;
  unsigned k = (lane < 16u) ? 0u : 8u;
  const bf16_t* p = src + (size_t)n * ld + k;
  v8bf lo = *(const v8bf*)(p);
  v8bf hi = *(const v8bf*)(p + 16);
  return __builtin_shufflevector(lo, hi, 0,1,2,3,4,5,6,7,8,9,10,11,12,13,14,15);
}

// D (16x16 f32): lane L holds column L%16, rows (L<16?0..7:8..15) across the 8 VGPRs.
__device__ inline void wmma_store_d_f32(float* dst, int ld, v8f c) {
  unsigned lane = threadIdx.x & 31u;
  unsigned n = lane & 15u;
  unsigned r0 = (lane < 16u) ? 0u : 8u;
#pragma unroll
  for (int j = 0; j < 8; j++) dst[(size_t)(r0 + j) * ld + n] = c[j];
}
__device__ inline void wmma_store_d_bf16(bf16_t* dst, int ld, v8f c) {
  unsigned lane = threadIdx.x & 31u;
  unsigned n = lane & 15u;
  unsigned r0 = (lane < 16u) ? 0u : 8u;
#pragma unroll
  for (int j = 0; j < 8; j++) dst[(size_t)(r0 + j) * ld + n] = (bf16_t)c[j];
}

// ---------------- K0a: pack all weights to bf16 ----------------
// wpk: [g(128x256) | theta(128x256) | phi(128x256)], wwbf: [256x128]
__global__ void k_pack(const float* __restrict__ gw, const float* __restrict__ tw,
                       const float* __restrict__ pw, const float* __restrict__ ww,
                       bf16_t* __restrict__ wpk, bf16_t* __restrict__ wwbf) {
  int i = blockIdx.x * 256 + threadIdx.x;
  if (i < CI_ * C_) {
    wpk[i]                = (bf16_t)gw[i];
    wpk[CI_ * C_ + i]     = (bf16_t)tw[i];
    wpk[2 * CI_ * C_ + i] = (bf16_t)pw[i];
    wwbf[i]               = (bf16_t)ww[i];
  }
}

// ---------------- K0b: transpose x [B,C,N] f32 -> xT [B,N,C] bf16 ----------------
__global__ void k_transpose(const float* __restrict__ x, bf16_t* __restrict__ xT) {
  __shared__ float tile[32][33];
  int b = blockIdx.z;
  int c0 = blockIdx.y * 32, s0 = blockIdx.x * 32;
  for (int i = threadIdx.y; i < 32; i += 8)
    tile[i][threadIdx.x] = x[((size_t)b * C_ + c0 + i) * HW_ + s0 + threadIdx.x];
  __syncthreads();
  for (int i = threadIdx.y; i < 32; i += 8)
    xT[((size_t)b * HW_ + s0 + i) * C_ + c0 + threadIdx.x] = (bf16_t)tile[threadIdx.x][i];
}

// ---------------- K1: three 1x1 convs as NT GEMMs (WMMA) ----------------
// Out[N,Ci] = xT[N,C] @ W[Ci,C]^T + bias, stored bf16 row-major.
__global__ void __launch_bounds__(128)
k_conv3(const bf16_t* __restrict__ xT, const bf16_t* __restrict__ wpk,
        const float* __restrict__ gb, const float* __restrict__ tb,
        const float* __restrict__ pb,
        bf16_t* __restrict__ gf, bf16_t* __restrict__ tf, bf16_t* __restrict__ pf) {
  int wsel = blockIdx.y;                 // 0=g, 1=theta, 2=phi
  int b = blockIdx.x / (HW_ / 32);
  int st = blockIdx.x % (HW_ / 32);      // 32-row spatial tile
  int w = threadIdx.x >> 5;              // wave -> 32 of the 128 Ci columns

  const bf16_t* A  = xT + ((size_t)b * HW_ + st * 32) * C_;
  const bf16_t* Bt = wpk + (size_t)wsel * CI_ * C_ + (size_t)(w * 32) * C_;
  const float* bias = (wsel == 0) ? gb : ((wsel == 1) ? tb : pb);
  bf16_t* out = ((wsel == 0) ? gf : ((wsel == 1) ? tf : pf)) + ((size_t)b * HW_ + st * 32) * CI_;

  v8f acc[2][2] = {};
  for (int k = 0; k < C_; k += 32) {
    v16bf a0 = wmma_load_a(A + k, C_);
    v16bf a1 = wmma_load_a(A + 16 * C_ + k, C_);
    v16bf b0 = wmma_load_b(Bt + k, C_);
    v16bf b1 = wmma_load_b(Bt + 16 * C_ + k, C_);
    acc[0][0] = WMMA(a0, b0, acc[0][0]);
    acc[0][1] = WMMA(a0, b1, acc[0][1]);
    acc[1][0] = WMMA(a1, b0, acc[1][0]);
    acc[1][1] = WMMA(a1, b1, acc[1][1]);
  }
  unsigned lane = threadIdx.x & 31u, n = lane & 15u, r0 = (lane < 16u) ? 0u : 8u;
#pragma unroll
  for (int rh = 0; rh < 2; rh++)
#pragma unroll
    for (int cg = 0; cg < 2; cg++) {
      float bv = bias[w * 32 + cg * 16 + n];
#pragma unroll
      for (int j = 0; j < 8; j++)
        out[(size_t)(rh * 16 + r0 + j) * CI_ + w * 32 + cg * 16 + n] =
            (bf16_t)(acc[rh][cg][j] + bv);
    }
}

// ---------------- K2: 3x3/2 maxpool; phi -> [B,Mp,Ci], g -> transposed [B,Ci,Mp] ----------------
__global__ void k_pool(const bf16_t* __restrict__ gf, const bf16_t* __restrict__ pf,
                       bf16_t* __restrict__ g_pT, bf16_t* __restrict__ phi_p) {
  int ci = threadIdx.x;      // 128
  int m  = blockIdx.x;       // 0..MP_-1
  int b  = blockIdx.y;
  float mg = 0.f, mp = 0.f;
  if (m < M_) {
    mg = -3.0e38f; mp = -3.0e38f;
    int py = m / PW_, px = m % PW_;
    int sy = py * 2, sx = px * 2;
#pragma unroll
    for (int dy = 0; dy < 3; dy++)
#pragma unroll
      for (int dx = 0; dx < 3; dx++) {
        int s = (sy + dy) * HH_ + sx + dx;
        size_t idx = ((size_t)b * HW_ + s) * CI_ + ci;
        mg = fmaxf(mg, (float)gf[idx]);
        mp = fmaxf(mp, (float)pf[idx]);
      }
  }
  phi_p[((size_t)b * MP_ + m) * CI_ + ci] = (bf16_t)mp;
  g_pT[((size_t)b * CI_ + ci) * MP_ + m] = (bf16_t)mg;
}

// ---------------- K3: fused attention: S = theta@phi^T (WMMA) -> exact softmax in LDS -> y = P@g (WMMA) ----------------
__global__ void __launch_bounds__(128)
k_attn(const bf16_t* __restrict__ tf, const bf16_t* __restrict__ phi_p,
       const bf16_t* __restrict__ g_pT, bf16_t* __restrict__ ybf) {
  extern __shared__ char smem[];
  float*  S    = (float*)smem;                               // 16 x MP_ f32
  bf16_t* Ast  = (bf16_t*)(smem + 16 * MP_ * 4);             // 16 x 128 bf16
  bf16_t* Pst  = Ast + 16 * CI_;                             // 16 x 32 bf16
  float*  rowM = (float*)(Pst + 16 * 32);
  float*  rowL = rowM + 16;

  int nt = blockIdx.x;   // 16-row block of N
  int b  = blockIdx.y;
  int w  = threadIdx.x >> 5;

  const bf16_t* theta = tf + ((size_t)b * HW_ + nt * 16) * CI_;

  // Stage theta row-block (16x128 bf16 = 4KB) into LDS.
#if USE_ASYNC_LDS
  {
    int t = threadIdx.x;
    const bf16_t* src = theta + t * 16;
    bf16_t* dst = Ast + t * 16;
    __builtin_amdgcn_global_load_async_to_lds_b128(TO_AS1(src),     TO_AS3(dst),     0, 0);
    __builtin_amdgcn_global_load_async_to_lds_b128(TO_AS1(src + 8), TO_AS3(dst + 8), 0, 0);
    wait_async0();
  }
#else
  {
    int t = threadIdx.x;
    *(v8bf*)(Ast + t * 16)     = *(const v8bf*)(theta + t * 16);
    *(v8bf*)(Ast + t * 16 + 8) = *(const v8bf*)(theta + t * 16 + 8);
  }
#endif
  __syncthreads();

  v16bf afrag[4];
#pragma unroll
  for (int kk = 0; kk < 4; kk++) afrag[kk] = wmma_load_a(Ast + kk * 32, CI_);

  // Pass 1: full score row-block into LDS (waves split M-tiles of 32).
  const bf16_t* phiB = phi_p + (size_t)b * MP_ * CI_;
  for (int t = w; t < MP_ / 32; t += 4) {
    int m0 = t * 32;
#pragma unroll
    for (int cg = 0; cg < 2; cg++) {
      v8f acc = {};
      const bf16_t* Bt = phiB + (size_t)(m0 + cg * 16) * CI_;
#pragma unroll
      for (int kk = 0; kk < 4; kk++)
        acc = WMMA(afrag[kk], wmma_load_b(Bt + kk * 32, CI_), acc);
      wmma_store_d_f32(S + m0 + cg * 16, MP_, acc);
    }
  }
  __syncthreads();

  // Row max / sum (8 threads per row, exact softmax over valid M columns).
  {
    int r = threadIdx.x >> 3, sub = threadIdx.x & 7;
    float mx = -3.0e38f;
    for (int c = sub; c < M_; c += 8) mx = fmaxf(mx, S[(size_t)r * MP_ + c]);
    for (int o = 1; o < 8; o <<= 1) mx = fmaxf(mx, __shfl_xor(mx, o, 32));
    float sm = 0.f;
    for (int c = sub; c < M_; c += 8) sm += __expf(S[(size_t)r * MP_ + c] - mx);
    for (int o = 1; o < 8; o <<= 1) sm += __shfl_xor(sm, o, 32);
    if (sub == 0) { rowM[r] = mx; rowL[r] = 1.0f / sm; }
  }
  __syncthreads();

  // Pass 2: y = P @ g, streaming P through a 16x32 bf16 staging tile.
  v8f yacc[2] = {};
  const bf16_t* gB = g_pT + (size_t)b * CI_ * MP_ + (size_t)(w * 32) * MP_;
  for (int t = 0; t < MP_ / 32; t++) {
    int m0 = t * 32;
    {
      int e0 = threadIdx.x * 4;
#pragma unroll
      for (int i = 0; i < 4; i++) {
        int e = e0 + i, r = e >> 5, c = e & 31;
        float v = (m0 + c < M_)
                    ? __expf(S[(size_t)r * MP_ + m0 + c] - rowM[r]) * rowL[r]
                    : 0.f;
        Pst[r * 32 + c] = (bf16_t)v;
      }
    }
    __syncthreads();
    v16bf pa = wmma_load_a(Pst, 32);
    yacc[0] = WMMA(pa, wmma_load_b(gB + m0, MP_), yacc[0]);
    yacc[1] = WMMA(pa, wmma_load_b(gB + (size_t)16 * MP_ + m0, MP_), yacc[1]);
    __syncthreads();
  }

  bf16_t* yo = ybf + ((size_t)b * HW_ + nt * 16) * CI_ + w * 32;
  wmma_store_d_bf16(yo, CI_, yacc[0]);
  wmma_store_d_bf16(yo + 16, CI_, yacc[1]);
}

// ---------------- K4: Out[C,N] = w_w[C,Ci] @ y[N,Ci]^T + bias + x (residual), f32 out ----------------
__global__ void __launch_bounds__(128)
k_final(const bf16_t* __restrict__ ybf, const bf16_t* __restrict__ wwbf,
        const float* __restrict__ wb, const float* __restrict__ x,
        float* __restrict__ out) {
  int nb = blockIdx.x;           // 128-col N tile
  int cb = blockIdx.y;           // 32-row C tile
  int b  = blockIdx.z;
  int w  = threadIdx.x >> 5;
  int n0 = nb * 128 + w * 32;
  int c0 = cb * 32;

  const bf16_t* A  = wwbf + (size_t)c0 * CI_;
  const bf16_t* Bt = ybf + ((size_t)b * HW_ + n0) * CI_;

  v8f acc[2][2] = {};
  for (int k = 0; k < CI_; k += 32) {
    v16bf a0 = wmma_load_a(A + k, CI_);
    v16bf a1 = wmma_load_a(A + 16 * CI_ + k, CI_);
    v16bf b0 = wmma_load_b(Bt + k, CI_);
    v16bf b1 = wmma_load_b(Bt + 16 * CI_ + k, CI_);
    acc[0][0] = WMMA(a0, b0, acc[0][0]);
    acc[0][1] = WMMA(a0, b1, acc[0][1]);
    acc[1][0] = WMMA(a1, b0, acc[1][0]);
    acc[1][1] = WMMA(a1, b1, acc[1][1]);
  }
  unsigned lane = threadIdx.x & 31u, n = lane & 15u, r0 = (lane < 16u) ? 0u : 8u;
#pragma unroll
  for (int rh = 0; rh < 2; rh++)
#pragma unroll
    for (int cg = 0; cg < 2; cg++)
#pragma unroll
      for (int j = 0; j < 8; j++) {
        int row = c0 + rh * 16 + r0 + j;
        int col = n0 + cg * 16 + n;
        size_t xi = ((size_t)b * C_ + row) * HW_ + col;
        out[xi] = acc[rh][cg][j] + wb[row] + x[xi];
      }
}

// ---------------- launcher ----------------
extern "C" void kernel_launch(void* const* d_in, const int* in_sizes, int n_in,
                              void* d_out, int out_size, void* d_ws, size_t ws_size,
                              hipStream_t stream) {
  (void)in_sizes; (void)n_in; (void)out_size; (void)ws_size;
  const float* x   = (const float*)d_in[0];
  const float* g_w = (const float*)d_in[1];
  const float* g_b = (const float*)d_in[2];
  const float* t_w = (const float*)d_in[3];
  const float* t_b = (const float*)d_in[4];
  const float* p_w = (const float*)d_in[5];
  const float* p_b = (const float*)d_in[6];
  const float* w_w = (const float*)d_in[7];
  const float* w_b = (const float*)d_in[8];
  float* out = (float*)d_out;

  char* ws = (char*)d_ws;
  size_t off = 0;
  auto take = [&](size_t bytes) -> char* {
    char* p = ws + off;
    off = (off + bytes + 255) & ~(size_t)255;
    return p;
  };
  bf16_t* xT    = (bf16_t*)take((size_t)B_ * HW_ * C_ * 2);   // 18.9 MB
  bf16_t* wpk   = (bf16_t*)take((size_t)3 * CI_ * C_ * 2);
  bf16_t* wwbf  = (bf16_t*)take((size_t)C_ * CI_ * 2);
  bf16_t* gf    = (bf16_t*)take((size_t)B_ * HW_ * CI_ * 2);  // 9.4 MB
  bf16_t* tf    = (bf16_t*)take((size_t)B_ * HW_ * CI_ * 2);
  bf16_t* pf    = (bf16_t*)take((size_t)B_ * HW_ * CI_ * 2);
  bf16_t* phi_p = (bf16_t*)take((size_t)B_ * MP_ * CI_ * 2);  // 2.3 MB
  bf16_t* g_pT  = (bf16_t*)take((size_t)B_ * CI_ * MP_ * 2);
  bf16_t* ybf   = (bf16_t*)take((size_t)B_ * HW_ * CI_ * 2);

  (void)hipFuncSetAttribute(reinterpret_cast<const void*>(k_attn),
                            hipFuncAttributeMaxDynamicSharedMemorySize, ATTN_LDS);

  k_pack<<<(CI_ * C_ + 255) / 256, 256, 0, stream>>>(g_w, t_w, p_w, w_w, wpk, wwbf);
  k_transpose<<<dim3(HW_ / 32, C_ / 32, B_), dim3(32, 8), 0, stream>>>(x, xT);
  k_conv3<<<dim3(B_ * (HW_ / 32), 3), 128, 0, stream>>>(xT, wpk, g_b, t_b, p_b, gf, tf, pf);
  k_pool<<<dim3(MP_, B_), CI_, 0, stream>>>(gf, pf, g_pT, phi_p);
  k_attn<<<dim3(NT16, B_), 128, ATTN_LDS, stream>>>(tf, phi_p, g_pT, ybf);
  k_final<<<dim3(HW_ / 128, C_ / 32, B_), 128, 0, stream>>>(ybf, wwbf, w_b, x, out);
}
